// ReparamLargeKernelConv_214748365445
// MI455X (gfx1250) — matrix operand order repaired and checked by reading, not compiled, not run on hardware
//
#include <hip/hip_runtime.h>
#include <hip/hip_bf16.h>

// ---- types ----
typedef __attribute__((ext_vector_type(16))) __bf16          v16bf;
typedef __attribute__((ext_vector_type(8)))  float           v8f;
typedef __attribute__((ext_vector_type(16))) unsigned short  v16u;
typedef __attribute__((ext_vector_type(8)))  unsigned short  v8u;

#define C_CH   384
#define HW     56
#define IMG    (HW*HW)          // 3136
#define NBATCH 16
#define KLK    31
#define PAD    15
#define LDS_W  120              // padded row stride (bf16 elems), 46 rows
#define LDS_CC 112              // staged valid columns: w in [-15, 96]

__device__ __forceinline__ unsigned short f32_to_bf16_rne(float f) {
    unsigned int b = __builtin_bit_cast(unsigned int, f);
    unsigned int r = b + 0x7FFFu + ((b >> 16) & 1u);
    return (unsigned short)(r >> 16);
}

// ---------------------------------------------------------------------------
// Prep: fold BN into weights, merge 5x5 into center of 31x31, build per-lane
// Toeplitz B fragments (contiguous K per lane half, same order as A) and per-channel bias.
// B frag for (c,kh,chunk): B[j,n] = W[kh, j-n], j = chunk*32 + K, n = lane&15.
// Lane l<16 holds K=0..15, lane l>=16 holds K=16..31 (16 bf16 per lane).
// ---------------------------------------------------------------------------
__global__ __launch_bounds__(32)
void prep_bfrag_bias(const float* __restrict__ lk_w,
                     const float* __restrict__ lk_g, const float* __restrict__ lk_b,
                     const float* __restrict__ lk_m, const float* __restrict__ lk_v,
                     const float* __restrict__ sk_w,
                     const float* __restrict__ sk_g, const float* __restrict__ sk_b,
                     const float* __restrict__ sk_m, const float* __restrict__ sk_v,
                     float* __restrict__ bias, unsigned short* __restrict__ bfrag) {
    const int c = blockIdx.x;
    const int l = threadIdx.x;

    const float s1 = lk_g[c] * rsqrtf(lk_v[c] + 1e-5f);
    const float s2 = sk_g[c] * rsqrtf(sk_v[c] + 1e-5f);
    if (l == 0)
        bias[c] = (lk_b[c] - lk_m[c] * s1) + (sk_b[c] - sk_m[c] * s2);

    __shared__ float Wc[KLK * KLK];
    for (int idx = l; idx < KLK * KLK; idx += 32) {
        const int kh = idx / KLK, kw = idx - kh * KLK;
        float w = s1 * lk_w[c * (KLK * KLK) + idx];
        if (kh >= 13 && kh < 18 && kw >= 13 && kw < 18)
            w += s2 * sk_w[c * 25 + (kh - 13) * 5 + (kw - 13)];
        Wc[idx] = w;
    }
    __syncthreads();

    const int n  = l & 15;
    const int kb = (l < 16) ? 0 : 16;
    for (int kh = 0; kh < KLK; ++kh) {
        for (int ch = 0; ch < 2; ++ch) {
            v16u pack;
            #pragma unroll
            for (int i = 0; i < 16; ++i) {
                const int j  = ch * 32 + kb + i;
                const int kw = j - n;
                const float w = (kw >= 0 && kw <= 30) ? Wc[kh * KLK + kw] : 0.0f;
                pack[i] = f32_to_bf16_rne(w);
            }
            *(v16u*)(bfrag + (((c * KLK + kh) * 2 + ch) * 32 + l) * 16) = pack;
        }
    }
}

// ---------------------------------------------------------------------------
// Main: one block = (n, c, h-tile). 4 waves, each wave owns one 16-wide w-tile.
// Stage zero-padded bf16 input strip in LDS; per kh do 2x wmma_f32_16x16x32_bf16.
// ---------------------------------------------------------------------------
__global__ __launch_bounds__(128)
void dwconv31_wmma(const float* __restrict__ x,
                   const float* __restrict__ bias,
                   const unsigned short* __restrict__ bfrag,
                   float* __restrict__ out) {
    const int b  = blockIdx.x;
    const int ht = b & 3;
    const int cn = b >> 2;
    const int c  = cn % C_CH;
    const int n  = cn / C_CH;
    const int h0 = ht * 16;

    __shared__ __align__(16) unsigned short s_in[46 * LDS_W];

    // ---- stage: rows r=0..45 (h = h0+r-15), cols cc=0..111 (w = cc-15) ----
    const float* __restrict__ xim = x + (size_t)(n * C_CH + c) * IMG;
    for (int idx = threadIdx.x; idx < 46 * LDS_CC; idx += 128) {
        const int r  = idx / LDS_CC;
        const int cc = idx - r * LDS_CC;
        const int h  = h0 + r - PAD;
        const int w  = cc - PAD;
        float v = 0.0f;
        if ((unsigned)h < (unsigned)HW && (unsigned)w < (unsigned)HW)
            v = xim[h * HW + w];
        s_in[r * LDS_W + cc] = f32_to_bf16_rne(v);
    }
    __syncthreads();

    const int l    = threadIdx.x & 31;
    const int wave = threadIdx.x >> 5;
    const int w0   = wave * 16;

    const int mrow = l & 15;              // A-matrix row this lane holds
    const int kbA  = (l < 16) ? 0 : 16;   // A K base (contiguous K order, same as B)

    const float bv = bias[c];
    v8f acc = {bv, bv, bv, bv, bv, bv, bv, bv};

    const unsigned short* __restrict__ bf_lane =
        bfrag + ((size_t)c * KLK) * 2 * 512 + (size_t)l * 16;

    for (int kh = 0; kh < KLK; ++kh) {
        const int rowoff = (kh + mrow) * LDS_W;
        #pragma unroll
        for (int ch = 0; ch < 2; ++ch) {
            // A fragment: lane holds K run {kbA..kbA+15}
            const int base = rowoff + w0 + ch * 32 + kbA;
            v8u alo = *(const v8u*)&s_in[base];
            v8u ahi = *(const v8u*)&s_in[base + 8];
            v16u a_u = __builtin_shufflevector(alo, ahi,
                0, 1, 2, 3, 4, 5, 6, 7, 8, 9, 10, 11, 12, 13, 14, 15);
            v16bf a = __builtin_bit_cast(v16bf, a_u);

            // B fragment: precomputed in exact lane layout, 32B coalesced
            v16u b_u = *(const v16u*)(bf_lane + (size_t)(kh * 2 + ch) * 512);
            v16bf bb = __builtin_bit_cast(v16bf, b_u);

            acc = __builtin_amdgcn_wmma_f32_16x16x32_bf16(
                false, a, false, bb, (short)0, acc, false, false);
        }
    }

    // ---- epilogue: f32 C/D layout: VGPR r -> M=r (lanes 0-15), M=r+8 (16-31)
    const int col   = w0 + (l & 15);
    const int rbase = h0 + ((l >= 16) ? 8 : 0);
    if (col < HW) {
        float* __restrict__ oim = out + (size_t)(n * C_CH + c) * IMG;
        #pragma unroll
        for (int r = 0; r < 8; ++r) {
            const int row = rbase + r;
            if (row < HW) oim[row * HW + col] = acc[r];
        }
    }
}

// ---------------------------------------------------------------------------
extern "C" void kernel_launch(void* const* d_in, const int* in_sizes, int n_in,
                              void* d_out, int out_size, void* d_ws, size_t ws_size,
                              hipStream_t stream) {
    const float* x     = (const float*)d_in[0];
    const float* lk_w  = (const float*)d_in[1];
    const float* lk_g  = (const float*)d_in[2];
    const float* lk_b  = (const float*)d_in[3];
    const float* lk_m  = (const float*)d_in[4];
    const float* lk_v  = (const float*)d_in[5];
    const float* sk_w  = (const float*)d_in[6];
    const float* sk_g  = (const float*)d_in[7];
    const float* sk_b  = (const float*)d_in[8];
    const float* sk_m  = (const float*)d_in[9];
    const float* sk_v  = (const float*)d_in[10];
    float* out = (float*)d_out;

    float* bias = (float*)d_ws;                                   // 384 floats
    unsigned short* bfrag = (unsigned short*)((char*)d_ws + 4096); // 384*31*2*512 u16 (~24.4MB)

    prep_bfrag_bias<<<C_CH, 32, 0, stream>>>(lk_w, lk_g, lk_b, lk_m, lk_v,
                                             sk_w, sk_g, sk_b, sk_m, sk_v,
                                             bias, bfrag);

    const int nblocks = NBATCH * C_CH * 4;   // (n, c, h-tile)
    dwconv31_wmma<<<nblocks, 128, 0, stream>>>(x, bias, bfrag, out);
}
